// Model_5265629905491
// MI455X (gfx1250) — matrix operand order repaired and checked by reading, not compiled, not run on hardware
//
#include <hip/hip_runtime.h>
#include <math.h>

#define BB 2
#define CC 128
#define HH 60
#define WW 108
#define HWN (HH*WW)            // 6480
#define RR 3
#define TAPS 49
#define LVLS 4
#define HID 192
#define CDIM 112
#define CORR_DIM 256
#define FLOW_DIM 64
#define CORR_CH (2*LVLS*TAPS)  // 392

typedef __attribute__((ext_vector_type(16))) __bf16 v16bf;
typedef __attribute__((ext_vector_type(8)))  float  v8f;

__device__ __forceinline__ float sigmoidf_(float x){ return 1.0f/(1.0f + __expf(-x)); }

// ---------------------------------------------------------------------------
// 2x2 average pooling (VALID) over trailing (H,W); planes = B*C
// ---------------------------------------------------------------------------
__global__ void pool2_kernel(const float* __restrict__ src, float* __restrict__ dst,
                             int planes, int Hs, int Ws){
  int Hd = Hs >> 1, Wd = Ws >> 1;
  int total = planes * Hd * Wd;
  int i = blockIdx.x * blockDim.x + threadIdx.x;
  if (i >= total) return;
  int xd = i % Wd; int t = i / Wd; int yd = t % Hd; int p = t / Hd;
  const float* s = src + (size_t)p * Hs * Ws;
  int ys = yd * 2, xs = xd * 2;
  float v = s[ys*Ws + xs] + s[ys*Ws + xs + 1] + s[(ys+1)*Ws + xs] + s[(ys+1)*Ws + xs + 1];
  dst[i] = 0.25f * v;
}

// ---------------------------------------------------------------------------
// Fused bidirectional correlation pyramid lookup (pyramid never materialized:
// avg-pool and bilinear sampling commute with the feature dot product).
// ---------------------------------------------------------------------------
__global__ void corr_lookup_kernel(
    const float* __restrict__ f0,   const float* __restrict__ f1,
    const float* __restrict__ p0l1, const float* __restrict__ p0l2, const float* __restrict__ p0l3,
    const float* __restrict__ p1l1, const float* __restrict__ p1l2, const float* __restrict__ p1l3,
    const float* __restrict__ flow0, const float* __restrict__ flow1,
    const float* __restrict__ embt, float* __restrict__ corr)
{
  int idx = blockIdx.x * blockDim.x + threadIdx.x;
  const int total = BB * CORR_CH * HWN;
  if (idx >= total) return;
  int n   = idx % HWN;
  int ch  = (idx / HWN) % CORR_CH;
  int b   = idx / (HWN * CORR_CH);
  int dir = ch / (LVLS * TAPS);
  int sub = ch % (LVLS * TAPS);
  int lvl = sub / TAPS;
  int tap = sub % TAPS;
  float dx = (float)(tap % 7 - RR);
  float dy = (float)(tap / 7 - RR);
  int px = n % WW, py = n / WW;
  float e = embt[b];
  float scale = (dir == 0) ? (1.0f / e) : (1.0f / (1.0f - e));
  const float* fl = (dir == 0) ? flow1 : flow0;
  float fx = fl[((size_t)b*2 + 0)*HWN + n];
  float fy = fl[((size_t)b*2 + 1)*HWN + n];
  float inv = 1.0f / (float)(1 << lvl);
  float cx = ((float)px + fx * scale) * inv + dx;
  float cy = ((float)py + fy * scale) * inv + dy;

  const int HL[4] = {60, 30, 15, 7};
  const int WL[4] = {108, 54, 27, 13};
  int Hl = HL[lvl], Wl = WL[lvl];

  const float* A;
  const float* img;
  if (dir == 0){
    A = f0 + (size_t)b * CC * HWN;
    img = (lvl == 0) ? f1   + (size_t)b*CC*HWN
        : (lvl == 1) ? p1l1 + (size_t)b*CC*(30*54)
        : (lvl == 2) ? p1l2 + (size_t)b*CC*(15*27)
        :              p1l3 + (size_t)b*CC*(7*13);
  } else {
    A = f1 + (size_t)b * CC * HWN;
    img = (lvl == 0) ? f0   + (size_t)b*CC*HWN
        : (lvl == 1) ? p0l1 + (size_t)b*CC*(30*54)
        : (lvl == 2) ? p0l2 + (size_t)b*CC*(15*27)
        :              p0l3 + (size_t)b*CC*(7*13);
  }

  float x0f = floorf(cx), y0f = floorf(cy);
  float wx = cx - x0f, wy = cy - y0f;
  int ix0 = (int)x0f, iy0 = (int)y0f;
  int ix1 = ix0 + 1,  iy1 = iy0 + 1;
  bool vx0 = (ix0 >= 0) && (ix0 < Wl), vx1 = (ix1 >= 0) && (ix1 < Wl);
  bool vy0 = (iy0 >= 0) && (iy0 < Hl), vy1 = (iy1 >= 0) && (iy1 < Hl);
  int cx0 = min(max(ix0, 0), Wl - 1), cx1 = min(max(ix1, 0), Wl - 1);
  int cy0 = min(max(iy0, 0), Hl - 1), cy1 = min(max(iy1, 0), Hl - 1);
  int o00 = cy0*Wl + cx0, o01 = cy0*Wl + cx1, o10 = cy1*Wl + cx0, o11 = cy1*Wl + cx1;
  float w00 = (1.0f-wx)*(1.0f-wy), w01 = wx*(1.0f-wy), w10 = (1.0f-wx)*wy, w11 = wx*wy;
  bool k00 = vx0 && vy0, k01 = vx1 && vy0, k10 = vx0 && vy1, k11 = vx1 && vy1;

  int plane = Hl * Wl;
  float acc = 0.0f;
  for (int c = 0; c < CC; ++c){
    const float* ic = img + (size_t)c * plane;
    float v00 = k00 ? ic[o00] : 0.0f;
    float v01 = k01 ? ic[o01] : 0.0f;
    float v10 = k10 ? ic[o10] : 0.0f;
    float v11 = k11 ? ic[o11] : 0.0f;
    float blend = w00*v00 + w01*v01 + w10*v10 + w11*v11;
    acc = fmaf(A[(size_t)c*HWN + n], blend, acc);
  }
  corr[idx] = acc * 0.08838834764831845f;  // 1/sqrt(128)
}

// ---------------------------------------------------------------------------
// Pack conv weights (f32, [Cout, Ktot]) into bf16 WMMA A-fragments:
// [octile][kchunk][lane(32)][half(16)] -> one contiguous coalesced 32B
// operand load per lane in the conv kernel.
// A 16x32 16-bit layout: lane L -> M = L&15; half e2=(j,h):
//   K = e2 + 16*(L>>4) (contiguous k order, same as the B fragment)
// ---------------------------------------------------------------------------
__global__ void pack_weights_kernel(const float* __restrict__ wgt, int Cout, int Ktot,
                                    __bf16* __restrict__ wp, int octiles, int kchunks){
  int total = octiles * kchunks * 32 * 16;
  int i = blockIdx.x * blockDim.x + threadIdx.x;
  if (i >= total) return;
  int e2   = i & 15;
  int lane = (i >> 4) & 31;
  int kc   = (i >> 9) % kchunks;
  int oct  = i / (kchunks << 9);
  int j = e2 >> 1, h = e2 & 1;
  int KlocA = e2 + ((lane >> 4) << 4);
  int kk = kc * 32 + KlocA;
  int oc = oct * 16 + (lane & 15);
  float v = (oc < Cout && kk < Ktot) ? wgt[(size_t)oc * Ktot + kk] : 0.0f;
  wp[i] = (__bf16)v;
}

// ---------------------------------------------------------------------------
// Implicit-GEMM conv (KS=1 or 3, zero pad), bf16 WMMA, f32 accumulate.
// Block = 128 threads = 4 waves; block tile = 8 oc-tiles x 64 pixels.
// Double-buffered LDS patch (stage kc+1 while WMMAs consume kc; one barrier
// per k-chunk). Per chunk a block runs 32 WMMAs; each wave reuses each of its 4
// B fragments across 2 oc-tiles.
// ---------------------------------------------------------------------------
#define CONV_BLOCK 128
template<int KS>
__global__ __launch_bounds__(CONV_BLOCK)
void conv_wmma_kernel(
    const __bf16* __restrict__ wpack, int kchunks,
    const float* __restrict__ x0, int c0, const float* __restrict__ g0,
    const float* __restrict__ x1, int c1,
    const float* __restrict__ x2, int c2,
    const float* __restrict__ bias,
    float* __restrict__ out, int Cout, size_t outBatchStride, int outChOff,
    int act)
{
  __shared__ __attribute__((aligned(32))) __bf16 sPatch[2][64 * 32];  // [buf][n][k]
  int b    = blockIdx.z;
  int n0   = blockIdx.x * 64;
  int wave = threadIdx.x >> 5;
  int lane = threadIdx.x & 31;
  int octiles = (Cout + 15) >> 4;
  int oct0 = (blockIdx.y * 4 + wave) * 2;
  int oct1 = oct0 + 1;
  int oct0c = min(oct0, octiles - 1);   // clamp: always-valid loads, masked stores
  int oct1c = min(oct1, octiles - 1);
  int cinTot = c0 + c1 + c2;
  int Ktot   = cinTot * KS * KS;
  const int pad = KS >> 1;
  int laneN  = lane & 15;
  int hi8    = (lane >> 4) << 3;
  int kbase  = (lane >> 4) << 4;        // B fragment: K = e + 16*(lane>>4)

  // staging assignment: fixed pixel per thread, k strided by 2
  int nl_sta = threadIdx.x & 63;
  int k_sta  = threadIdx.x >> 6;        // 0 or 1
  int nn_sta = n0 + nl_sta;
  int py_sta = nn_sta / WW, px_sta = nn_sta % WW;
  bool npix_ok = nn_sta < HWN;

  const float* xb0 = x0 ? x0 + (size_t)b * c0 * HWN : nullptr;
  const float* gb0 = g0 ? g0 + (size_t)b * c0 * HWN : nullptr;
  const float* xb1 = x1 ? x1 + (size_t)b * c1 * HWN : nullptr;
  const float* xb2 = x2 ? x2 + (size_t)b * c2 * HWN : nullptr;

  auto stage = [&](int kc, int buf){
    int kb = kc * 32;
    #pragma unroll 4
    for (int it = 0; it < 16; ++it){
      int k  = k_sta + it * 2;
      int kk = kb + k;
      float v = 0.0f;
      if (npix_ok && kk < Ktot){
        int icg = kk / (KS * KS);           // constant divisor -> mul/shift
        int r   = kk - icg * (KS * KS);
        int ky  = r / KS;
        int kx  = r - ky * KS;
        int yy  = py_sta + ky - pad;
        int xx  = px_sta + kx - pad;
        if (yy >= 0 && yy < HH && xx >= 0 && xx < WW){
          int nidx = yy * WW + xx;
          if (icg < c0){
            v = xb0[(size_t)icg * HWN + nidx];
            if (gb0) v *= gb0[(size_t)icg * HWN + nidx];
          } else if (icg < c0 + c1){
            v = xb1[(size_t)(icg - c0) * HWN + nidx];
          } else {
            v = xb2[(size_t)(icg - c0 - c1) * HWN + nidx];
          }
        }
      }
      sPatch[buf][nl_sta * 32 + k] = (__bf16)v;
    }
  };

  v8f acc[2][4];
  #pragma unroll
  for (int oi = 0; oi < 2; ++oi)
    #pragma unroll
    for (int t = 0; t < 4; ++t)
      #pragma unroll
      for (int j = 0; j < 8; ++j) acc[oi][t][j] = 0.0f;

  stage(0, 0);
  __syncthreads();

  for (int kc = 0; kc < kchunks; ++kc){
    int cur = kc & 1;
    // stage next chunk into the other buffer (safe: its readers passed the
    // barrier at the end of the previous iteration)
    if (kc + 1 < kchunks) stage(kc + 1, cur ^ 1);

    // pre-packed A fragments: 1 contiguous 32B load per lane per oc-tile
    v16bf a0 = *(const v16bf*)(wpack + (((size_t)oct0c * kchunks + kc) * 32 + lane) * 16);
    v16bf a1 = *(const v16bf*)(wpack + (((size_t)oct1c * kchunks + kc) * 32 + lane) * 16);

    #pragma unroll
    for (int t = 0; t < 4; ++t){
      v16bf bt = *(const v16bf*)(&sPatch[cur][(t * 16 + laneN) * 32 + kbase]);
      acc[0][t] = __builtin_amdgcn_wmma_f32_16x16x32_bf16(
          false, a0, false, bt, (short)0, acc[0][t], false, false);
      acc[1][t] = __builtin_amdgcn_wmma_f32_16x16x32_bf16(
          false, a1, false, bt, (short)0, acc[1][t], false, false);
    }
    __syncthreads();
  }

  // epilogue: bias + activation + masked store
  #pragma unroll
  for (int oi = 0; oi < 2; ++oi){
    int oct = oct0 + oi;
    if (oct >= octiles) continue;
    #pragma unroll
    for (int t = 0; t < 4; ++t){
      int n = n0 + t * 16 + laneN;
      if (n >= HWN) continue;
      #pragma unroll
      for (int j = 0; j < 8; ++j){
        int oc = oct * 16 + j + hi8;
        if (oc < Cout){
          float v = acc[oi][t][j] + bias[oc];
          if      (act == 1) v = fmaxf(v, 0.0f);
          else if (act == 2) v = sigmoidf_(v);
          else if (act == 3) v = tanhf(v);
          out[(size_t)b * outBatchStride + (size_t)(outChOff + oc) * HWN + n] = v;
        }
      }
    }
  }
}

// ---------------------------------------------------------------------------
// GRU state blend: h' = (1-z)*h + z*q
// ---------------------------------------------------------------------------
__global__ void gru_update_kernel(const float* __restrict__ h, const float* __restrict__ z,
                                  const float* __restrict__ q, float* __restrict__ hn, int total){
  int i = blockIdx.x * blockDim.x + threadIdx.x;
  if (i >= total) return;
  float zz = z[i];
  hn[i] = (1.0f - zz) * h[i] + zz * q[i];
}

// ---------------------------------------------------------------------------
extern "C" void kernel_launch(void* const* d_in, const int* in_sizes, int n_in,
                              void* d_out, int out_size, void* d_ws, size_t ws_size,
                              hipStream_t stream) {
  (void)in_sizes; (void)n_in; (void)out_size; (void)ws_size;
  const float* fmap0 = (const float*)d_in[0];
  const float* fmap1 = (const float*)d_in[1];
  const float* ft    = (const float*)d_in[2];
  const float* flow0 = (const float*)d_in[3];
  const float* flow1 = (const float*)d_in[4];
  const float* embt  = (const float*)d_in[5];
  const float* w_corr = (const float*)d_in[6];  const float* b_corr = (const float*)d_in[7];
  const float* w_flow = (const float*)d_in[8];  const float* b_flow = (const float*)d_in[9];
  const float* w_ctx  = (const float*)d_in[10]; const float* b_ctx  = (const float*)d_in[11];
  const float* w_z    = (const float*)d_in[12]; const float* b_z    = (const float*)d_in[13];
  const float* w_r    = (const float*)d_in[14]; const float* b_r    = (const float*)d_in[15];
  const float* w_q    = (const float*)d_in[16]; const float* b_q    = (const float*)d_in[17];
  const float* w_fh   = (const float*)d_in[18]; const float* b_fh   = (const float*)d_in[19];
  const float* w_dh   = (const float*)d_in[20]; const float* b_dh   = (const float*)d_in[21];
  float* outp = (float*)d_out;

  // Workspace (float units; every region is a multiple of 8 floats -> 32B aligned)
  float* ws = (float*)d_ws;
  size_t off = 0;
  auto alloc = [&](size_t nE){ float* p = ws + off; off += nE; return p; };
  const size_t szL1 = (size_t)BB*CC*30*54, szL2 = (size_t)BB*CC*15*27, szL3 = (size_t)BB*CC*7*13;
  float* p1l1 = alloc(szL1); float* p1l2 = alloc(szL2); float* p1l3 = alloc(szL3);
  float* p0l1 = alloc(szL1); float* p0l2 = alloc(szL2); float* p0l3 = alloc(szL3);
  float* corr  = alloc((size_t)BB*CORR_CH*HWN);
  float* cfeat = alloc((size_t)BB*CORR_DIM*HWN);
  float* ffeat = alloc((size_t)BB*FLOW_DIM*HWN);
  float* hctx  = alloc((size_t)BB*HID*HWN);
  float* zb    = alloc((size_t)BB*HID*HWN);
  float* rb    = alloc((size_t)BB*HID*HWN);
  float* qb    = alloc((size_t)BB*HID*HWN);
  float* hn    = alloc((size_t)BB*HID*HWN);
  // bf16 weight-fragment regions (elem counts are multiples of 512)
  auto allocbf = [&](size_t nElems){ __bf16* p = (__bf16*)(ws + off); off += nElems / 2; return p; };

  struct ConvCfg { const float* w; int Cout; int Ktot; int octiles; int kchunks; __bf16* wp; };
  auto mkcfg = [&](const float* w, int Cout, int Ktot){
    ConvCfg c; c.w = w; c.Cout = Cout; c.Ktot = Ktot;
    c.octiles = (Cout + 15) / 16; c.kchunks = (Ktot + 31) / 32;
    c.wp = allocbf((size_t)c.octiles * c.kchunks * 512);
    return c;
  };
  ConvCfg cCorr = mkcfg(w_corr, CORR_DIM, CORR_CH * 1);
  ConvCfg cFlow = mkcfg(w_flow, FLOW_DIM, 4 * 9);
  ConvCfg cCtx  = mkcfg(w_ctx,  HID,  CDIM * 9);
  ConvCfg cZ    = mkcfg(w_z,    HID,  (HID + CORR_DIM + FLOW_DIM) * 9);
  ConvCfg cR    = mkcfg(w_r,    HID,  (HID + CORR_DIM + FLOW_DIM) * 9);
  ConvCfg cQ    = mkcfg(w_q,    HID,  (HID + CORR_DIM + FLOW_DIM) * 9);
  ConvCfg cFh   = mkcfg(w_fh,   4,    HID * 9);
  ConvCfg cDh   = mkcfg(w_dh,   CDIM, HID * 9);

  auto pack = [&](const ConvCfg& c){
    int total = c.octiles * c.kchunks * 512;
    pack_weights_kernel<<<(total + 255)/256, 256, 0, stream>>>(
        c.w, c.Cout, c.Ktot, c.wp, c.octiles, c.kchunks);
  };
  pack(cCorr); pack(cFlow); pack(cCtx); pack(cZ); pack(cR); pack(cQ); pack(cFh); pack(cDh);

  // 1. Pooled feature pyramids (pooling commutes with correlation).
  auto pool = [&](const float* src, float* dst, int Hs, int Ws){
    int planes = BB * CC;
    int total = planes * (Hs/2) * (Ws/2);
    pool2_kernel<<<(total + 255)/256, 256, 0, stream>>>(src, dst, planes, Hs, Ws);
  };
  pool(fmap1, p1l1, 60, 108); pool(p1l1, p1l2, 30, 54); pool(p1l2, p1l3, 15, 27);
  pool(fmap0, p0l1, 60, 108); pool(p0l1, p0l2, 30, 54); pool(p0l2, p0l3, 15, 27);

  // 2. Fused bidirectional pyramid lookup -> corr [B, 392, H, W]
  {
    int total = BB * CORR_CH * HWN;
    corr_lookup_kernel<<<(total + 255)/256, 256, 0, stream>>>(
        fmap0, fmap1, p0l1, p0l2, p0l3, p1l1, p1l2, p1l3, flow0, flow1, embt, corr);
  }

  // 3..: WMMA conv stack
  auto conv = [&](const ConvCfg& c,
                  const float* x0, int c0, const float* g0,
                  const float* x1, int c1, const float* x2, int c2,
                  const float* bias,
                  float* out, size_t obs, int ocoff, int ks, int act){
    dim3 grid((HWN + 63) / 64, (c.octiles + 7) / 8, BB);
    if (ks == 1)
      conv_wmma_kernel<1><<<grid, CONV_BLOCK, 0, stream>>>(
          c.wp, c.kchunks, x0, c0, g0, x1, c1, x2, c2, bias,
          out, c.Cout, obs, ocoff, act);
    else
      conv_wmma_kernel<3><<<grid, CONV_BLOCK, 0, stream>>>(
          c.wp, c.kchunks, x0, c0, g0, x1, c1, x2, c2, bias,
          out, c.Cout, obs, ocoff, act);
  };

  // motion encoder
  conv(cCorr, corr, CORR_CH, nullptr, nullptr, 0, nullptr, 0, b_corr,
       cfeat, (size_t)CORR_DIM*HWN, 0, 1, /*relu*/1);
  conv(cFlow, flow0, 2, nullptr, flow1, 2, nullptr, 0, b_flow,
       ffeat, (size_t)FLOW_DIM*HWN, 0, 3, /*relu*/1);
  // context
  conv(cCtx, ft, CDIM, nullptr, nullptr, 0, nullptr, 0, b_ctx,
       hctx, (size_t)HID*HWN, 0, 3, /*tanh*/3);
  // ConvGRU gates: hx = [h | cfeat | ffeat]
  conv(cZ, hctx, HID, nullptr, cfeat, CORR_DIM, ffeat, FLOW_DIM, b_z,
       zb, (size_t)HID*HWN, 0, 3, /*sigmoid*/2);
  conv(cR, hctx, HID, nullptr, cfeat, CORR_DIM, ffeat, FLOW_DIM, b_r,
       rb, (size_t)HID*HWN, 0, 3, /*sigmoid*/2);
  // q: input = [r*h | cfeat | ffeat]
  conv(cQ, hctx, HID, rb, cfeat, CORR_DIM, ffeat, FLOW_DIM, b_q,
       qb, (size_t)HID*HWN, 0, 3, /*tanh*/3);
  // h' = (1-z)h + z q
  {
    int total = BB * HID * HWN;
    gru_update_kernel<<<(total + 255)/256, 256, 0, stream>>>(hctx, zb, qb, hn, total);
  }
  // heads -> out [B, 116, H, W]: delta_ft (0..111), delta_flow (112..115)
  conv(cDh, hn, HID, nullptr, nullptr, 0, nullptr, 0, b_dh,
       outp, (size_t)116*HWN, 0, 3, 0);
  conv(cFh, hn, HID, nullptr, nullptr, 0, nullptr, 0, b_fh,
       outp, (size_t)116*HWN, 112, 3, 0);
}
